// TaskAlignedAssigner_2078764171610
// MI455X (gfx1250) — compile-verified, matrix operand order checked by reading, not run on hardware
//
#include <hip/hip_runtime.h>
#include <math.h>

// ---- problem constants (from setup_inputs) ----
#define BB   8
#define LL   8400
#define NN   32
#define CC   80
#define TOPK 13
#define EPSF 1e-9f
#define MAXV 8

typedef __attribute__((ext_vector_type(16))) _Float16 v16h;
typedef __attribute__((ext_vector_type(8)))  float    v8f;

// ---------------- geometry helpers ----------------
__device__ __forceinline__ void rcorners(float cx, float cy, float w, float h,
                                         float ang, float* X, float* Y) {
  float c = cosf(ang), s = sinf(ang);
  const float dx[4] = {-0.5f * w, 0.5f * w, 0.5f * w, -0.5f * w};
  const float dy[4] = {-0.5f * h, -0.5f * h, 0.5f * h, 0.5f * h};
#pragma unroll
  for (int k = 0; k < 4; ++k) {
    X[k] = cx + dx[k] * c - dy[k] * s;
    Y[k] = cy + dx[k] * s + dy[k] * c;
  }
}

// Sutherland–Hodgman intersection area, reproducing the reference's
// padded-array (pad == first vertex), roll(-1) wraparound, and den clamp.
__device__ float inter_area(const float* x1, const float* y1,
                            const float* x2, const float* y2) {
  float px[MAXV], py[MAXV];
#pragma unroll
  for (int i = 0; i < 4; ++i) { px[i] = x1[i]; py[i] = y1[i]; }
#pragma unroll
  for (int i = 4; i < MAXV; ++i) { px[i] = x1[0]; py[i] = y1[0]; }
  int cnt = 4;
#pragma unroll
  for (int e = 0; e < 4; ++e) {
    float p1x = x2[e], p1y = y2[e];
    float p2x = x2[(e + 1) & 3], p2y = y2[(e + 1) & 3];
    float dx = p2x - p1x, dy = p2y - p1y;
    float s[MAXV];
#pragma unroll
    for (int i = 0; i < MAXV; ++i)
      s[i] = dx * (py[i] - p1y) - dy * (px[i] - p1x);
    float ox[MAXV], oy[MAXV];
    int m = 0;
    for (int i = 0; i < MAXV; ++i) {
      if (i < cnt) {
        int j = (i + 1) & (MAXV - 1);
        bool ic = s[i] >= 0.0f, in = s[j] >= 0.0f;
        if (ic) { if (m < MAXV) { ox[m] = px[i]; oy[m] = py[i]; } m++; }
        if (ic != in) {
          float den = s[i] - s[j];
          if (fabsf(den) <= 1e-12f) den = 1e-12f;
          float t = s[i] / den;
          if (m < MAXV) {
            ox[m] = px[i] + t * (px[j] - px[i]);
            oy[m] = py[i] + t * (py[j] - py[i]);
          }
          m++;
        }
      }
    }
    cnt = m > MAXV ? MAXV : m;
    if (cnt == 0) {
      for (int i = 0; i < MAXV; ++i) { px[i] = 0.0f; py[i] = 0.0f; }
    } else {
      for (int i = 0; i < MAXV; ++i) {
        px[i] = (i < cnt) ? ox[i] : ox[0];
        py[i] = (i < cnt) ? oy[i] : oy[0];
      }
    }
  }
  float area = 0.0f;
#pragma unroll
  for (int i = 0; i < MAXV; ++i) {
    int j = (i + 1) & (MAXV - 1);
    area += px[i] * py[j] - px[j] * py[i];
  }
  return 0.5f * fabsf(area);
}

// ---------------- K1: per-pair metrics ----------------
__global__ __launch_bounds__(256) void k1_metrics(
    const float* __restrict__ pd_scores, const float* __restrict__ pd_bboxes,
    const float* __restrict__ anc, const int* __restrict__ gt_labels,
    const float* __restrict__ gt_bboxes, const float* __restrict__ mask_gt,
    float* __restrict__ g_align, float* __restrict__ g_ov,
    unsigned char* __restrict__ g_mv) {
  int bn = blockIdx.y;           // b*NN + n
  int b = bn >> 5;
  int l = blockIdx.x * blockDim.x + threadIdx.x;
  if (l >= LL) return;

  const float* gb = gt_bboxes + bn * 5;
  float gx = gb[0], gy = gb[1], gw = gb[2], gh = gb[3], ga = gb[4];
  int lab = gt_labels[bn];
  bool gvalid = mask_gt[bn] > 0.0f;

  float X1[4], Y1[4];
  rcorners(gx, gy, gw, gh, ga, X1, Y1);

  // point-in-rotated-box
  float pxp = anc[2 * l], pyp = anc[2 * l + 1];
  float abx = X1[1] - X1[0], aby = Y1[1] - Y1[0];
  float adx = X1[3] - X1[0], ady = Y1[3] - Y1[0];
  float apx = pxp - X1[0], apy = pyp - Y1[0];
  float nab = abx * abx + aby * aby, nad = adx * adx + ady * ady;
  float pab = apx * abx + apy * aby, pdd = apx * adx + apy * ady;
  bool inb = (pab >= 0.0f) && (pab <= nab) && (pdd >= 0.0f) && (pdd <= nad);
  bool mv = inb && gvalid;

  const float* pb = pd_bboxes + ((size_t)b * LL + l) * 5;
  __builtin_prefetch(pb, 0, 0);  // global_prefetch_b8
  float X2[4], Y2[4];
  rcorners(pb[0], pb[1], pb[2], pb[3], pb[4], X2, Y2);

  float inter = inter_area(X1, Y1, X2, Y2);
  float a1 = fabsf(gw * gh), a2 = fabsf(pb[2] * pb[3]);
  float iou = inter / (a1 + a2 - inter + EPSF);
  float sc = pd_scores[((size_t)b * LL + l) * CC + lab];

  size_t ix = (size_t)bn * LL + l;
  float ov = mv ? iou : 0.0f;
  g_ov[ix] = ov;
  g_align[ix] = mv ? sc * sc * ov : 0.0f;
  g_mv[ix] = mv ? 1 : 0;
}

// ---------------- K2: per-GT top-13 -> mask_pos ----------------
__global__ __launch_bounds__(256) void k2_topk(
    const float* __restrict__ g_align, const unsigned char* __restrict__ g_mv,
    const float* __restrict__ mask_gt, float* __restrict__ g_mp) {
  __shared__ float sv[256];
  __shared__ int si[256];
  __shared__ int sel[TOPK];
  int bn = blockIdx.x;
  int tid = threadIdx.x;
  const float* row = g_align + (size_t)bn * LL;
  const unsigned char* mvrow = g_mv + (size_t)bn * LL;
  float* mprow = g_mp + (size_t)bn * LL;

  for (int l = tid; l < LL; l += 256) mprow[l] = 0.0f;
  __syncthreads();
  if (!(mask_gt[bn] > 0.0f)) return;  // invalid gt: row stays zero

  for (int k = 0; k < TOPK; ++k) {
    float bv = -1.0f;
    int bi = LL;
    for (int l = tid; l < LL; l += 256) {
      bool skip = false;
      for (int kk = 0; kk < k; ++kk) if (sel[kk] == l) skip = true;
      if (!skip) {
        float v = row[l];
        if (v > bv) { bv = v; bi = l; }  // strict > keeps lowest index on ties
      }
    }
    sv[tid] = bv; si[tid] = bi;
    __syncthreads();
    for (int s = 128; s > 0; s >>= 1) {
      if (tid < s) {
        if (sv[tid + s] > sv[tid] ||
            (sv[tid + s] == sv[tid] && si[tid + s] < si[tid])) {
          sv[tid] = sv[tid + s]; si[tid] = si[tid + s];
        }
      }
      __syncthreads();
    }
    if (tid == 0) sel[k] = si[0];
    __syncthreads();
  }
  if (tid < TOPK) {
    int l = sel[tid];
    mprow[l] = mvrow[l] ? 1.0f : 0.0f;  // mask_topk * mask_valid
  }
}

// ---------------- K3: WMMA fg + conflict resolution + gathers ----------------
// fg[col] = sum_n mask_pos[n][col] via V_WMMA_F32_16X16X32_F16 with A = ones.
// B order here (contiguous, fine with A = ones): lanes 0-15 hold K=0-15, lanes 16-31 hold K=16-31,
// VGPR r packs K=2r,2r+1. With all-ones A the column sum is exact for 0/1 data.
__global__ __launch_bounds__(256) void k3_resolve(
    const int* __restrict__ gt_labels, const float* __restrict__ gt_bboxes,
    const float* __restrict__ g_ov, float* __restrict__ g_mp,
    float* __restrict__ o_lab, float* __restrict__ o_box,
    float* __restrict__ o_fg, float* __restrict__ o_idx,
    int* __restrict__ g_tlab) {
  int tid = threadIdx.x;
  int wave = tid >> 5;
  int lane = tid & 31;
  int gw = blockIdx.x * 8 + wave;     // 0 .. B*L/16 - 1 (4200 waves)
  int abase = gw * 16;                // 16 anchors per wave, same batch (L%16==0)
  int b = abase / LL;
  int l0 = abase % LL;
  int col = lane & 15;
  int hi = lane >> 4;
  int l = l0 + col;

  const float* mpB = g_mp + (size_t)b * NN * LL;

  v16h bf;
#pragma unroll
  for (int j = 0; j < 16; ++j) {
    int n = hi * 16 + j;              // lanes 0-15: K=0-15; lanes 16-31: K=16-31
    bf[j] = (_Float16)mpB[(size_t)n * LL + l];
  }
  v16h af;
#pragma unroll
  for (int j = 0; j < 16; ++j) af[j] = (_Float16)1.0f;
  v8f c = {};
  v8f d = __builtin_amdgcn_wmma_f32_16x16x32_f16(
      /*neg_a=*/false, af, /*neg_b=*/false, bf,
      /*c_mod=*/(short)0, c, /*reuse_a=*/false, /*reuse_b=*/false);
  float fg = d[0];                    // every lane: fg of its own column

  if (hi == 0) {
    int fgc = (int)(fg + 0.5f);
    int idx = 0;
    float fgm;
    const float* ovB = g_ov + (size_t)b * NN * LL;
    float* mpW = g_mp + (size_t)b * NN * LL;
    if (fgc > 1) {
      float best = -1.0f;
      for (int n = 0; n < NN; ++n) {
        float ov = ovB[(size_t)n * LL + l];
        if (ov > best) { best = ov; idx = n; }   // first max, like jnp.argmax
      }
      for (int n = 0; n < NN; ++n)
        mpW[(size_t)n * LL + l] = (n == idx) ? 1.0f : 0.0f;
      fgm = 1.0f;
    } else {
      fgm = (float)fgc;
      if (fgc >= 1) {
        for (int n = 0; n < NN; ++n) {
          if (mpW[(size_t)n * LL + l] > 0.5f) { idx = n; break; }
        }
      }
    }
    int gi = b * NN + idx;
    size_t t = (size_t)b * LL + l;
    o_lab[t] = (float)gt_labels[gi];
    g_tlab[t] = gt_labels[gi];
#pragma unroll
    for (int k = 0; k < 5; ++k) o_box[t * 5 + k] = gt_bboxes[gi * 5 + k];
    o_fg[t] = (fgm > 0.0f) ? 1.0f : 0.0f;
    o_idx[t] = (float)idx;
  }
}

// ---------------- K4: pos_align[b,n] = max_l align*mask_pos ----------------
__global__ __launch_bounds__(256) void k4_posalign(
    const float* __restrict__ g_align, const float* __restrict__ g_mp,
    float* __restrict__ g_pa) {
  __shared__ float sm[256];
  int bn = blockIdx.x;
  int tid = threadIdx.x;
  size_t base = (size_t)bn * LL;
  float m = 0.0f;                      // products are non-negative
  for (int l = tid; l < LL; l += 256)
    m = fmaxf(m, g_align[base + l] * g_mp[base + l]);
  sm[tid] = m;
  __syncthreads();
  for (int s = 128; s > 0; s >>= 1) {
    if (tid < s) sm[tid] = fmaxf(sm[tid], sm[tid + s]);
    __syncthreads();
  }
  if (tid == 0) g_pa[bn] = sm[0];
}

// ---------------- K5: norm + one-hot target_scores ----------------
__global__ __launch_bounds__(256) void k5_scores(
    const float* __restrict__ g_align, const float* __restrict__ g_mp,
    const float* __restrict__ g_pa, const int* __restrict__ g_tlab,
    const float* __restrict__ o_fg, float* __restrict__ o_sc) {
  int t = blockIdx.x * blockDim.x + threadIdx.x;
  if (t >= BB * LL) return;
  int b = t / LL;
  int l = t - b * LL;
  float nrm = 0.0f;                    // ratios are non-negative
  for (int n = 0; n < NN; ++n) {
    size_t ix = ((size_t)(b * NN + n)) * LL + l;
    float v = g_align[ix] * g_mp[ix];
    nrm = fmaxf(nrm, v / (g_pa[b * NN + n] + EPSF));
  }
  bool fg = o_fg[t] > 0.0f;
  int lab = g_tlab[t];
  float* sc = o_sc + (size_t)t * CC;
  for (int cc = 0; cc < CC; ++cc) sc[cc] = 0.0f;
  if (fg) sc[lab] = nrm;
}

// ---------------- launch ----------------
extern "C" void kernel_launch(void* const* d_in, const int* in_sizes, int n_in,
                              void* d_out, int out_size, void* d_ws,
                              size_t ws_size, hipStream_t stream) {
  (void)in_sizes; (void)n_in; (void)out_size; (void)ws_size;
  const float* pd_scores = (const float*)d_in[0];   // (B,L,C)
  const float* pd_bboxes = (const float*)d_in[1];   // (B,L,5)
  const float* anc       = (const float*)d_in[2];   // (L,2)
  const int*   gt_labels = (const int*)d_in[3];     // (B,N,1)
  const float* gt_bboxes = (const float*)d_in[4];   // (B,N,5)
  const float* mask_gt   = (const float*)d_in[5];   // (B,N,1)

  const size_t BL  = (size_t)BB * LL;
  const size_t BNL = (size_t)BB * NN * LL;

  float* wsf = (float*)d_ws;
  float* g_align = wsf;                 // BNL f32
  float* g_ov    = wsf + BNL;           // BNL f32
  float* g_mp    = wsf + 2 * BNL;       // BNL f32
  float* g_pa    = wsf + 3 * BNL;       // B*N f32
  int*   g_tlab  = (int*)(wsf + 3 * BNL + BB * NN);      // BL i32
  unsigned char* g_mv = (unsigned char*)(g_tlab + BL);   // BNL u8

  float* o = (float*)d_out;             // tuple outputs concatenated (float)
  float* o_lab = o;                     // (B,L)
  float* o_box = o + BL;                // (B,L,5)
  float* o_sc  = o + BL * 6;            // (B,L,C)
  float* o_fg  = o + BL * (6 + CC);     // (B,L)
  float* o_idx = o + BL * (7 + CC);     // (B,L)

  dim3 g1((LL + 255) / 256, BB * NN);
  k1_metrics<<<g1, 256, 0, stream>>>(pd_scores, pd_bboxes, anc, gt_labels,
                                     gt_bboxes, mask_gt, g_align, g_ov, g_mv);
  k2_topk<<<BB * NN, 256, 0, stream>>>(g_align, g_mv, mask_gt, g_mp);
  k3_resolve<<<(BB * LL) / 128, 256, 0, stream>>>(gt_labels, gt_bboxes, g_ov,
                                                  g_mp, o_lab, o_box, o_fg,
                                                  o_idx, g_tlab);
  k4_posalign<<<BB * NN, 256, 0, stream>>>(g_align, g_mp, g_pa);
  k5_scores<<<(int)((BL + 255) / 256), 256, 0, stream>>>(g_align, g_mp, g_pa,
                                                         g_tlab, o_fg, o_sc);
}
